// GraphSAGEModel_57028575756303
// MI455X (gfx1250) — matrix operand order repaired and checked by reading, not compiled, not run on hardware
//
#include <hip/hip_runtime.h>
#include <hip/hip_bf16.h>
#include <math.h>

#define NNODES 100000
#define NEDGES 1600000

typedef __attribute__((ext_vector_type(16))) __bf16 v16bf;
typedef __attribute__((ext_vector_type(8)))  float  v8f;

union A16 { v16bf v; __bf16 e[16]; };
union F8  { v8f  v; float  e[8];  };

// ---------------------------------------------------------------- elementwise
__global__ void zero_f32(float* __restrict__ p, size_t n) {
    size_t i = (size_t)blockIdx.x * blockDim.x + threadIdx.x;
    if (i < n) p[i] = 0.0f;
}

__global__ void count_deg(const int* __restrict__ dst, float* __restrict__ cnt, int ne) {
    size_t i = (size_t)blockIdx.x * blockDim.x + threadIdx.x;
    if (i < (size_t)ne) atomicAdd(&cnt[dst[i]], 1.0f);
}

__global__ void recip_fin(float* __restrict__ r, int n) {
    size_t i = (size_t)blockIdx.x * blockDim.x + threadIdx.x;
    if (i < (size_t)n) r[i] = 1.0f / fmaxf(r[i], 1.0f);
}

// scatter-add first C channels of feat rows (row stride fstride) into out [*,C]
__global__ void scatter_add(const float* __restrict__ feat, const int* __restrict__ src,
                            const int* __restrict__ dst, float* __restrict__ out,
                            int C, int fstride, int ne) {
    size_t i = (size_t)blockIdx.x * blockDim.x + threadIdx.x;
    int per = C >> 2;
    size_t total = (size_t)ne * per;
    if (i >= total) return;
    int e  = (int)(i / per);
    int c4 = (int)(i % per) << 2;
    int s = src[e], d = dst[e];
    const float4 v = *(const float4*)(feat + (size_t)s * fstride + c4);
    float* o = out + (size_t)d * C + c4;
    atomicAdd(o + 0, v.x); atomicAdd(o + 1, v.y);
    atomicAdd(o + 2, v.z); atomicAdd(o + 3, v.w);
}

// ------------------------------------------------------------------- BN stats
__global__ void bn_stats(const float* __restrict__ h, float* __restrict__ sum,
                         float* __restrict__ sumsq, int n, int C) {
    int c   = threadIdx.x % C;
    int rpb = blockDim.x / C;
    int r0  = blockIdx.x * rpb + threadIdx.x / C;
    float s = 0.0f, ss = 0.0f;
    for (int r = r0; r < n; r += gridDim.x * rpb) {
        float v = h[(size_t)r * C + c];
        s += v; ss += v * v;
    }
    atomicAdd(&sum[c], s);
    atomicAdd(&sumsq[c], ss);
}

__global__ void bn_apply_relu(float* __restrict__ h, const float* __restrict__ sum,
                              const float* __restrict__ sumsq, const float* __restrict__ g,
                              const float* __restrict__ b, int n, int C) {
    size_t i = (size_t)blockIdx.x * blockDim.x + threadIdx.x;
    if (i >= (size_t)n * C) return;
    int c = (int)(i % C);
    float invn = 1.0f / (float)n;
    float m  = sum[c] * invn;
    float v  = sumsq[c] * invn - m * m;
    float sc = rsqrtf(v + 1e-5f) * g[c];
    float val = (h[i] - m) * sc + b[c];
    h[i] = val > 0.0f ? val : 0.0f;
}

// ------------------------------------------------- fused SAGE GEMM (WMMA bf16)
// One wave per 16-row strip. A panel ([agg*recip | x], K = 2*CIN) is loaded
// into registers ONCE, then the wave sweeps all COUT/16 column tiles,
// re-loading only the (L2-resident) 16x32 B tiles. 8x less A traffic than
// one-wave-per-tile.
template<int CIN, int COUT>
__global__ __launch_bounds__(32)
void sage_gemm(const float* __restrict__ agg, const float* __restrict__ xin,
               const float* __restrict__ recip,
               const float* __restrict__ Wl, const float* __restrict__ Wr,
               const float* __restrict__ bias, float* __restrict__ out) {
    constexpr int KSTEPS = (2 * CIN) / 32;
    const int lane = threadIdx.x & 31;
    const int half = lane >> 4;
    const int lr   = lane & 15;
    const int tm   = blockIdx.x;
    const int row  = tm * 16 + lr;

    const float* aggRow = agg + (size_t)row * CIN;
    const float* xRow   = xin + (size_t)row * CIN;
    const float  rs     = recip[row];

    // ---- load full A panel into registers (bf16) --------------------------
    A16 a[KSTEPS];
    #pragma unroll
    for (int ks = 0; ks < KSTEPS; ++ks) {
        const int k0 = ks * 32;
        int ck = k0 + 8 * half;                 // elements 0..7  : K = ck..ck+7
        const float* p; float s;
        if (ck < CIN) { p = aggRow + ck; s = rs; } else { p = xRow + (ck - CIN); s = 1.0f; }
        #pragma unroll
        for (int i = 0; i < 8; ++i) a[ks].e[i] = (__bf16)(p[i] * s);
        ck = k0 + 16 + 8 * half;                // elements 8..15 : K = ck..ck+7
        if (ck < CIN) { p = aggRow + ck; s = rs; } else { p = xRow + (ck - CIN); s = 1.0f; }
        #pragma unroll
        for (int i = 0; i < 8; ++i) a[ks].e[8 + i] = (__bf16)(p[i] * s);
    }

    // ---- sweep column tiles ----------------------------------------------
    #pragma unroll
    for (int tn = 0; tn < COUT / 16; ++tn) {
        const int wrow = tn * 16 + lr;          // output channel fed by this lane
        const float* WlRow = Wl + (size_t)wrow * CIN;
        const float* WrRow = Wr + (size_t)wrow * CIN;

        F8 acc;
        const float bv = bias[wrow];
        #pragma unroll
        for (int i = 0; i < 8; ++i) acc.e[i] = bv;

        #pragma unroll
        for (int ks = 0; ks < KSTEPS; ++ks) {
            int ck = ks * 32 + 8 * half;        // elements 0..7  : K = ck..ck+7
            const float* p = (ck < CIN) ? (WlRow + ck) : (WrRow + (ck - CIN));
            A16 b;
            #pragma unroll
            for (int i = 0; i < 8; ++i) b.e[i] = (__bf16)p[i];
            ck = ks * 32 + 16 + 8 * half;       // elements 8..15 : K = ck..ck+7
            p = (ck < CIN) ? (WlRow + ck) : (WrRow + (ck - CIN));
            #pragma unroll
            for (int i = 0; i < 8; ++i) b.e[8 + i] = (__bf16)p[i];
            acc.v = __builtin_amdgcn_wmma_f32_16x16x32_bf16(false, a[ks].v, false, b.v,
                                                            (short)0, acc.v, false, false);
        }
        // C/D layout: lane -> col = lane&15 ; element mv -> row = mv + 8*half
        #pragma unroll
        for (int mv = 0; mv < 8; ++mv) {
            int orow = tm * 16 + mv + 8 * half;
            out[(size_t)orow * COUT + tn * 16 + lr] = acc.e[mv];
        }
    }
}

// ----------------------------------------------------- plain X @ W^T (+bias)
template<int CIN, int COUT, bool RELU>
__global__ __launch_bounds__(32)
void gemm_xwt(const float* __restrict__ X, const float* __restrict__ W,
              const float* __restrict__ bias, float* __restrict__ out) {
    constexpr int KSTEPS = CIN / 32;
    const int lane = threadIdx.x & 31;
    const int half = lane >> 4;
    const int lr   = lane & 15;
    const int tm   = blockIdx.x;
    const int row  = tm * 16 + lr;

    const float* xRow = X + (size_t)row * CIN;

    A16 a[KSTEPS];
    #pragma unroll
    for (int ks = 0; ks < KSTEPS; ++ks) {
        const int k0 = ks * 32;
        #pragma unroll
        for (int i = 0; i < 8; ++i) a[ks].e[i]     = (__bf16)xRow[k0 + 8 * half + i];
        #pragma unroll
        for (int i = 0; i < 8; ++i) a[ks].e[8 + i] = (__bf16)xRow[k0 + 16 + 8 * half + i];
    }

    #pragma unroll
    for (int tn = 0; tn < COUT / 16; ++tn) {
        const int wrow = tn * 16 + lr;
        const float* wRow = W + (size_t)wrow * CIN;

        F8 acc;
        const float bv = bias ? bias[wrow] : 0.0f;
        #pragma unroll
        for (int i = 0; i < 8; ++i) acc.e[i] = bv;

        #pragma unroll
        for (int ks = 0; ks < KSTEPS; ++ks) {
            A16 b;
            #pragma unroll
            for (int i = 0; i < 8; ++i) b.e[i] = (__bf16)wRow[ks * 32 + 8 * half + i];
            #pragma unroll
            for (int i = 0; i < 8; ++i) b.e[8 + i] = (__bf16)wRow[ks * 32 + 16 + 8 * half + i];
            acc.v = __builtin_amdgcn_wmma_f32_16x16x32_bf16(false, a[ks].v, false, b.v,
                                                            (short)0, acc.v, false, false);
        }
        #pragma unroll
        for (int mv = 0; mv < 8; ++mv) {
            int orow = tm * 16 + mv + 8 * half;
            float v = acc.e[mv];
            if (RELU) v = v > 0.0f ? v : 0.0f;
            out[(size_t)orow * COUT + tn * 16 + lr] = v;
        }
    }
}

// layer-2 combine: h2 = aggb*recip + bl2 + t[:,64:128]
__global__ void combine_l2(const float* __restrict__ aggb, const float* __restrict__ recip,
                           const float* __restrict__ t, const float* __restrict__ bl2,
                           float* __restrict__ h2, int n) {
    size_t i = (size_t)blockIdx.x * blockDim.x + threadIdx.x;
    if (i >= (size_t)n * 64) return;
    int r = (int)(i >> 6), c = (int)(i & 63);
    h2[i] = aggb[i] * recip[r] + bl2[c] + t[(size_t)r * 128 + 64 + c];
}

// head: out = sigmoid(h3 @ Wc2^T + bc2)
__global__ void classifier_out(const float* __restrict__ h3, const float* __restrict__ Wc2,
                               const float* __restrict__ bc2, float* __restrict__ out, int n) {
    size_t i = (size_t)blockIdx.x * blockDim.x + threadIdx.x;
    if (i >= (size_t)n) return;
    const float4* hp = (const float4*)(h3 + i * 32);
    const float4* wp = (const float4*)Wc2;
    float z = bc2[0];
    #pragma unroll
    for (int j = 0; j < 8; ++j) {
        float4 hv = hp[j], wv = wp[j];
        z += hv.x * wv.x + hv.y * wv.y + hv.z * wv.z + hv.w * wv.w;
    }
    out[i] = 1.0f / (1.0f + expf(-z));
}

// =============================================================================
extern "C" void kernel_launch(void* const* d_in, const int* in_sizes, int n_in,
                              void* d_out, int out_size, void* d_ws, size_t ws_size,
                              hipStream_t stream) {
    const float* x   = (const float*)d_in[0];
    const int*   ei  = (const int*)d_in[1];
    const int* src = ei;
    const int* dst = ei + NEDGES;
    const float* Wl0 = (const float*)d_in[2];
    const float* bl0 = (const float*)d_in[3];
    const float* Wr0 = (const float*)d_in[4];
    const float* g0  = (const float*)d_in[5];
    const float* b0  = (const float*)d_in[6];
    const float* Wl1 = (const float*)d_in[7];
    const float* bl1 = (const float*)d_in[8];
    const float* Wr1 = (const float*)d_in[9];
    const float* g1  = (const float*)d_in[10];
    const float* b1  = (const float*)d_in[11];
    const float* Wl2 = (const float*)d_in[12];
    const float* bl2 = (const float*)d_in[13];
    const float* Wr2 = (const float*)d_in[14];
    const float* g2  = (const float*)d_in[15];
    const float* b2  = (const float*)d_in[16];
    const float* Wc1 = (const float*)d_in[17];
    const float* bc1 = (const float*)d_in[18];
    const float* Wc2 = (const float*)d_in[19];
    const float* bc2 = (const float*)d_in[20];

    // ---- workspace layout (bytes) -----------------------------------------
    char* ws = (char*)d_ws;
    const size_t MB = 1024ull * 1024ull;
    size_t o = 0;
    float* recip = (float*)(ws + o); o += 512 * 1024;      // N floats
    float* agg   = (float*)(ws + o); o += 52 * MB;         // up to N*128
    float* hB    = (float*)(ws + o); o += 52 * MB;         // h0, later t
    float* hC    = (float*)(ws + o); o += 52 * MB;         // h1
    float* h2    = (float*)(ws + o); o += 26 * MB;         // N*64
    float* h3    = (float*)(ws + o); o += 13 * MB;         // N*32
    float* W2cat = (float*)(ws + o); o += 128 * 1024;      // 128x128
    float* stats = (float*)(ws + o); o += 4 * 1024;        // sum|sumsq

    const int N = NNODES, E = NEDGES;
    auto nb = [](size_t n, int b) { return (unsigned)((n + (size_t)b - 1) / (size_t)b); };

    // ---- degrees ----------------------------------------------------------
    zero_f32<<<nb(N, 256), 256, 0, stream>>>(recip, (size_t)N);
    count_deg<<<nb(E, 256), 256, 0, stream>>>(dst, recip, E);
    recip_fin<<<nb(N, 256), 256, 0, stream>>>(recip, N);

    // ---- layer 0: 16 -> 128 ----------------------------------------------
    zero_f32<<<nb((size_t)N * 16, 256), 256, 0, stream>>>(agg, (size_t)N * 16);
    scatter_add<<<nb((size_t)E * 4, 256), 256, 0, stream>>>(x, src, dst, agg, 16, 16, E);
    sage_gemm<16, 128><<<N / 16, 32, 0, stream>>>(agg, x, recip, Wl0, Wr0, bl0, hB);
    zero_f32<<<1, 256, 0, stream>>>(stats, 256);
    bn_stats<<<1024, 256, 0, stream>>>(hB, stats, stats + 128, N, 128);
    bn_apply_relu<<<nb((size_t)N * 128, 256), 256, 0, stream>>>(hB, stats, stats + 128, g0, b0, N, 128);

    // ---- layer 1: 128 -> 128 ----------------------------------------------
    zero_f32<<<nb((size_t)N * 128, 256), 256, 0, stream>>>(agg, (size_t)N * 128);
    scatter_add<<<nb((size_t)E * 32, 256), 256, 0, stream>>>(hB, src, dst, agg, 128, 128, E);
    sage_gemm<128, 128><<<N / 16, 32, 0, stream>>>(agg, hB, recip, Wl1, Wr1, bl1, hC);
    zero_f32<<<1, 256, 0, stream>>>(stats, 256);
    bn_stats<<<1024, 256, 0, stream>>>(hC, stats, stats + 128, N, 128);
    bn_apply_relu<<<nb((size_t)N * 128, 256), 256, 0, stream>>>(hC, stats, stats + 128, g1, b1, N, 128);

    // ---- layer 2: 128 -> 64 (transform-then-aggregate; halves scatter) ----
    hipMemcpyAsync(W2cat, Wl2, 64 * 128 * sizeof(float), hipMemcpyDeviceToDevice, stream);
    hipMemcpyAsync(W2cat + 64 * 128, Wr2, 64 * 128 * sizeof(float), hipMemcpyDeviceToDevice, stream);
    gemm_xwt<128, 128, false><<<N / 16, 32, 0, stream>>>(hC, W2cat, nullptr, hB); // t = [h1@Wl2^T | h1@Wr2^T]
    zero_f32<<<nb((size_t)N * 64, 256), 256, 0, stream>>>(agg, (size_t)N * 64);
    scatter_add<<<nb((size_t)E * 16, 256), 256, 0, stream>>>(hB, src, dst, agg, 64, 128, E); // scatter t[:, :64]
    combine_l2<<<nb((size_t)N * 64, 256), 256, 0, stream>>>(agg, recip, hB, bl2, h2, N);
    zero_f32<<<1, 256, 0, stream>>>(stats, 256);
    bn_stats<<<1024, 256, 0, stream>>>(h2, stats, stats + 64, N, 64);
    bn_apply_relu<<<nb((size_t)N * 64, 256), 256, 0, stream>>>(h2, stats, stats + 64, g2, b2, N, 64);

    // ---- classifier -------------------------------------------------------
    gemm_xwt<64, 32, true><<<N / 16, 32, 0, stream>>>(h2, Wc1, bc1, h3);
    classifier_out<<<nb(N, 256), 256, 0, stream>>>(h3, Wc2, bc2, (float*)d_out, N);
}
